// KnowingFoam_15204184228362
// MI455X (gfx1250) — matrix operand order repaired and checked by reading, not compiled, not run on hardware
//
#include <hip/hip_runtime.h>
#include <hip/hip_bf16.h>

#define VOCAB 32000
#define DIM   256
#define NB    256
#define SEQ   64
#define NFR   3
#define THRESH 1.5f
#define EQ_FACTOR 0.59049f   // 0.9^5 closed form of 5 relaxation steps
#define NSWEEP 8
#define LW 8                 // waves per logits block

typedef __attribute__((ext_vector_type(16))) __bf16 v16bf;
typedef __attribute__((ext_vector_type(8)))  __bf16 v8bf;
typedef __attribute__((ext_vector_type(8)))  float  v8f;
typedef __attribute__((ext_vector_type(2)))  float  v2f;
typedef __attribute__((ext_vector_type(4)))  int    v4i;

#if __has_builtin(__builtin_amdgcn_global_load_async_to_lds_b32) && \
    __has_builtin(__builtin_amdgcn_global_load_async_to_lds_b128) && \
    __has_builtin(__builtin_amdgcn_s_wait_asynccnt)
#define HAVE_ASYNC_LDS 1
#else
#define HAVE_ASYNC_LDS 0
#endif

// async-load-to-LDS builtins take typed AS(1) global / AS(3) LDS pointers
#define GPTR32(p)  ((__attribute__((address_space(1))) int*)(p))
#define LPTR32(p)  ((__attribute__((address_space(3))) int*)(p))
#define GPTR128(p) ((__attribute__((address_space(1))) v4i*)(p))
#define LPTR128(p) ((__attribute__((address_space(3))) v4i*)(p))

// single dynamic-LDS symbol shared by the kernels that need big LDS
extern __shared__ char dyn_lds[];

// ---------------------------------------------------------------- init / convert

__global__ void init_kernel(float* memory, float* means, float* vars) {
    size_t i = (size_t)blockIdx.x * blockDim.x + threadIdx.x;
    if (i < (size_t)NB * DIM) memory[i] = 0.0f;
    if (i < (size_t)NFR * NB * DIM) { means[i] = 0.0f; vars[i] = 1.0f; }
}

__global__ void cvtE_kernel(const float* __restrict__ E, __bf16* __restrict__ Ebf, int n) {
    int i = blockIdx.x * blockDim.x + threadIdx.x;
    if (i < n) Ebf[i] = (__bf16)E[i];
}

// ---------------------------------------------------------------- step 1: novelty / decay / xm

__global__ __launch_bounds__(256)
void prep_kernel(const int* __restrict__ tokens, int step,
                 const float* __restrict__ E, const float* __restrict__ memory,
                 float* __restrict__ xm, float* __restrict__ decayp,
                 const float* __restrict__ mdb, const float* __restrict__ ns) {
    __shared__ float red[256];
    int d = threadIdx.x;
    float s = 0.0f;
    for (int n = 0; n < NB; ++n) s += memory[(size_t)n * DIM + d];
    float mm = s * (1.0f / NB);
    int tok = tokens[step];
    float x = E[(size_t)tok * DIM + d];

    auto blockReduce = [&](float v) -> float {
        red[d] = v; __syncthreads();
        for (int off = 128; off > 0; off >>= 1) {
            if (d < off) red[d] += red[d + off];
            __syncthreads();
        }
        float r = red[0]; __syncthreads();
        return r;
    };

    float dot  = blockReduce(x * mm);
    float xx   = blockReduce(x * x);
    float mmmm = blockReduce(mm * mm);

    float xn = sqrtf(xx) + 1e-10f;
    float mn = sqrtf(mmmm) + 1e-10f;
    float novelty = (mn > 1e-8f) ? (1.0f - dot / (xn * mn)) : 1.0f;
    float sens = fabsf(ns[0]);
    float dec = 1.0f / (1.0f + expf(-(mdb[0] - sens * novelty)));
    if (d == 0) decayp[0] = dec;
    xm[d] = x + dec * mm;
}

// ---------------------------------------------------------------- step 2: bubble measurements

__global__ __launch_bounds__(256)
void meas_kernel(const float* __restrict__ xm, const float* __restrict__ W,
                 float* __restrict__ meas) {
    __shared__ __attribute__((aligned(16))) float sxm[DIM];
    int e = threadIdx.x, n = blockIdx.x;
#if HAVE_ASYNC_LDS
    __builtin_amdgcn_global_load_async_to_lds_b32(GPTR32(xm + e), LPTR32(&sxm[e]), 0, 0);
    __builtin_amdgcn_s_wait_asynccnt(0);
#else
    sxm[e] = xm[e];
#endif
    __syncthreads();
    const float* Wn = W + (size_t)n * DIM * DIM;
    float acc = 0.0f;
    #pragma unroll 4
    for (int dd = 0; dd < DIM; ++dd) {
        if ((dd & 31) == 0 && dd + 32 < DIM)
            __builtin_prefetch(Wn + (size_t)(dd + 32) * DIM + e, 0, 0); // global_prefetch_b8
        acc += sxm[dd] * Wn[(size_t)dd * DIM + e];
    }
    meas[(size_t)n * DIM + e] = acc;
}

// ---------------------------------------------------------------- step 3: surprise (deterministic 2-stage)

__global__ __launch_bounds__(256)
void surpriseA_kernel(const float* __restrict__ meas, const float* __restrict__ means,
                      const float* __restrict__ vars, float* __restrict__ partials) {
    __shared__ float red[256];
    int f = blockIdx.y, chunk = blockIdx.x, t = threadIdx.x;
    size_t base = (size_t)chunk * 1024;
    const float* mf = means + (size_t)f * NB * DIM;
    const float* vf = vars  + (size_t)f * NB * DIM;
    float s = 0.0f;
    #pragma unroll
    for (int i = 0; i < 4; ++i) {
        size_t idx = base + t + (size_t)i * 256;
        s += fabsf(meas[idx] - mf[idx]) * rsqrtf(vf[idx] + 1e-8f);
    }
    red[t] = s; __syncthreads();
    for (int off = 128; off > 0; off >>= 1) {
        if (t < off) red[t] += red[t + off];
        __syncthreads();
    }
    if (t == 0) partials[f * 64 + chunk] = red[0];
}

__global__ __launch_bounds__(64)
void surpriseB_kernel(const float* __restrict__ partials, int step, int* __restrict__ flag) {
    __shared__ float red[64];
    __shared__ int know;
    int t = threadIdx.x;
    if (t == 0) know = 0;
    __syncthreads();
    for (int f = 0; f < NFR; ++f) {
        red[t] = partials[f * 64 + t];
        __syncthreads();
        for (int off = 32; off > 0; off >>= 1) {
            if (t < off) red[t] += red[t + off];
            __syncthreads();
        }
        if (t == 0) {
            float sur = red[0] / (float)(NB * DIM);
            if (sur < THRESH && step > 0) know = 1;
        }
        __syncthreads();
    }
    if (t == 0) flag[0] = know;
}

// ---------------------------------------------------------------- step 4: bubble mean + fused state update

__global__ __launch_bounds__(256)
void colmean_kernel(const float* __restrict__ meas, float* __restrict__ mu) {
    int dd = threadIdx.x;
    float s = 0.0f;
    for (int n = 0; n < NB; ++n) s += meas[(size_t)n * DIM + dd];
    mu[dd] = s * (1.0f / NB);
}

__global__ __launch_bounds__(256)
void state_kernel(const float* __restrict__ meas, const float* __restrict__ mu,
                  const int* __restrict__ flag, const float* __restrict__ decayp,
                  float* __restrict__ memory, float* __restrict__ means,
                  float* __restrict__ vars, float* __restrict__ eff,
                  float* __restrict__ mnorm, int step) {
    __shared__ float red[256];
    int dd = threadIdx.x, n = blockIdx.x;
    size_t i = (size_t)n * DIM + dd;
    float m = meas[i];
    float e = flag[0] ? m : (mu[dd] + EQ_FACTOR * (m - mu[dd]));  // closed-form equilibrate
    eff[i] = e;
    float dec = decayp[0];
    memory[i] = dec * memory[i] + (1.0f - dec) * e;
    const float drs[NFR] = {0.3f, 0.7f, 0.9f};
    #pragma unroll
    for (int f = 0; f < NFR; ++f) {
        size_t fi = (size_t)f * NB * DIM + i;
        float dr = drs[f];
        float mo = means[fi], vo = vars[fi];
        float mnew = dr * mo + (1.0f - dr) * e;
        float vnew = fmaxf(dr * vo + (1.0f - dr) * (e - mo) * (e - mnew), 1e-8f);
        if (step == 0) { mnew = e; vnew = 0.1f; }
        means[fi] = mnew; vars[fi] = vnew;
    }
    red[dd] = e * e; __syncthreads();
    for (int off = 128; off > 0; off >>= 1) {
        if (dd < off) red[dd] += red[dd + off];
        __syncthreads();
    }
    float nrm = sqrtf(red[0]) + 1e-10f;
    mnorm[i] = e / nrm;
}

// ---------------------------------------------------------------- step 5: rho = mnorm^T mnorm / NB  (WMMA f32 16x16x4)

__global__ __launch_bounds__(32)
void rho_kernel(const float* __restrict__ mnorm, float* __restrict__ rho,
                __bf16* __restrict__ rho_bf) {
    const int lane = threadIdx.x;
    const int mt = blockIdx.x >> 4, nt = blockIdx.x & 15;
    const int m0 = mt * 16, n0 = nt * 16;
    const bool hi = lane >= 16;
    const int l = lane & 15;
    v8f acc = {};
    for (int k0 = 0; k0 < NB; k0 += 4) {
        int ka = k0 + (hi ? 2 : 0);
        // A (16x4 f32): lanes 0-15 -> K=k0,k0+1 ; lanes 16-31 -> K=k0+2,k0+3 ; M = l
        v2f a, b;
        a.x = mnorm[(size_t)ka * DIM + m0 + l];
        a.y = mnorm[(size_t)(ka + 1) * DIM + m0 + l];
        // B (4x16 f32): same K split, N = l
        b.x = mnorm[(size_t)ka * DIM + n0 + l];
        b.y = mnorm[(size_t)(ka + 1) * DIM + n0 + l];
        acc = __builtin_amdgcn_wmma_f32_16x16x4_f32(false, a, false, b, (short)0, acc, false, false);
    }
    #pragma unroll
    for (int r = 0; r < 8; ++r) {
        int mrow = m0 + r + (hi ? 8 : 0);
        int ncol = n0 + l;
        float v = acc[r] * (1.0f / NB);
        rho[(size_t)mrow * DIM + ncol] = v;
        rho_bf[(size_t)mrow * DIM + ncol] = (__bf16)v;
    }
}

// ---------------------------------------------------------------- step 6: eigvalsh via parallel Jacobi, matrix in LDS
// CDNA5: 320KB LDS per WGP; matrix kept in dynamic LDS (256KB, size given at launch).

__global__ __launch_bounds__(256)
void eig_kernel(const float* __restrict__ rho, float* __restrict__ out) {
    float* A = (float*)dyn_lds;                       // DIM*DIM f32 = 256 KB dynamic LDS
    __shared__ int top[128], bot[128];
    __shared__ float cs[128], sn[128];
    __shared__ float sh[256];
    __shared__ float red[256];
    int t = threadIdx.x;
    // stage rho -> LDS (async path: 1 KB per thread, 64 x b128)
#if HAVE_ASYNC_LDS
    {
        const char* g = (const char*)rho;
        char* l = (char*)A;
        #pragma unroll 4
        for (int i = 0; i < 64; ++i) {
            int off = (t + i * 256) * 16;
            __builtin_amdgcn_global_load_async_to_lds_b128(GPTR128(g + off), LPTR128(l + off), 0, 0);
        }
        __builtin_amdgcn_s_wait_asynccnt(0);
    }
#else
    for (int r = 0; r < DIM; ++r) A[r * DIM + t] = rho[(size_t)r * DIM + t];
#endif
    if (t < 128) { top[t] = t; bot[t] = 255 - t; }
    __syncthreads();

    for (int round = 0; round < NSWEEP * (DIM - 1); ++round) {
        if (t < 128) {
            int p = top[t], q = bot[t];
            float app = A[p * DIM + p];
            float aqq = A[q * DIM + q];
            float apq = A[p * DIM + q];
            float c = 1.0f, s = 0.0f;
            if (fabsf(apq) > 1e-30f) {
                float tau = (aqq - app) / (2.0f * apq);
                float tt = (tau >= 0.0f ? 1.0f : -1.0f) / (fabsf(tau) + sqrtf(1.0f + tau * tau));
                c = rsqrtf(1.0f + tt * tt);
                s = tt * c;
            }
            cs[t] = c; sn[t] = s;
        }
        __syncthreads();
        // row rotations (pairs touch disjoint rows)
        for (int i = 0; i < 128; ++i) {
            int p = top[i], q = bot[i];
            float c = cs[i], s = sn[i];
            float ap = A[p * DIM + t], aq = A[q * DIM + t];
            A[p * DIM + t] = c * ap - s * aq;
            A[q * DIM + t] = s * ap + c * aq;
        }
        __syncthreads();
        // column rotations (pairs touch disjoint columns)
        for (int i = 0; i < 128; ++i) {
            int p = top[i], q = bot[i];
            float c = cs[i], s = sn[i];
            float ap = A[t * DIM + p], aq = A[t * DIM + q];
            A[t * DIM + p] = c * ap - s * aq;
            A[t * DIM + q] = s * ap + c * aq;
        }
        __syncthreads();
        if (t == 0) {   // round-robin tournament rotation, top[0] fixed
            int b0 = bot[0];
            for (int i = 0; i < 127; ++i) bot[i] = bot[i + 1];
            bot[127] = top[127];
            for (int i = 127; i >= 2; --i) top[i] = top[i - 1];
            top[1] = b0;
        }
        __syncthreads();
    }

    sh[t] = A[t * DIM + t];
    __syncthreads();
    // bitonic sort ascending (matches eigvalsh ordering)
    for (int k = 2; k <= 256; k <<= 1)
        for (int j = k >> 1; j > 0; j >>= 1) {
            int ixj = t ^ j;
            if (ixj > t) {
                float a = sh[t], b = sh[ixj];
                bool up = ((t & k) == 0);
                if ((a > b) == up) { sh[t] = b; sh[ixj] = a; }
            }
            __syncthreads();
        }
    float e = fmaxf(sh[t], 1e-12f);
    red[t] = e; __syncthreads();
    for (int off = 128; off > 0; off >>= 1) {
        if (t < off) red[t] += red[t + off];
        __syncthreads();
    }
    out[t] = e / red[0];
}

// ---------------------------------------------------------------- step 7: logits[v] = e_v^T rho e_v
// 8 waves/block; whole rho_bf16 (128KB dynamic LDS) async-staged once per block.

__global__ __launch_bounds__(256)
void logits_kernel(const __bf16* __restrict__ Ebf, const float* __restrict__ E,
                   const __bf16* __restrict__ rho_bf, float* __restrict__ out) {
    __bf16* srho = (__bf16*)dyn_lds;                  // DIM*DIM bf16 = 128 KB dynamic LDS
    const int t = threadIdx.x;
    {
        const char* g = (const char*)rho_bf;
        char* l = (char*)srho;
        #pragma unroll 4
        for (int i = 0; i < 32; ++i) {
            int off = (t + i * 256) * 16;   // 512 B per thread
#if HAVE_ASYNC_LDS
            __builtin_amdgcn_global_load_async_to_lds_b128(GPTR128(g + off), LPTR128(l + off), 0, 0);
#else
            *(float4*)(l + off) = *(const float4*)(g + off);
#endif
        }
#if HAVE_ASYNC_LDS
        __builtin_amdgcn_s_wait_asynccnt(0);
#endif
        __syncthreads();
    }

    const int lane = t & 31;
    const int wave = t >> 5;
    const int v0 = (blockIdx.x * LW + wave) * 16;
    const bool hi = lane >= 16;
    const int l = lane & 15;
    float rowacc[8];
    #pragma unroll
    for (int r = 0; r < 8; ++r) rowacc[r] = 0.0f;

    for (int et = 0; et < DIM / 16; ++et) {
        const int e0 = et * 16;
        v8f acc = {};
        #pragma unroll
        for (int kt = 0; kt < DIM / 32; ++kt) {
            const int k0 = kt * 32;
            // A 16x32 bf16: lanes 0-15 row l hold K=k0..k0+7 / k0+16..k0+23 ; hi lanes offset +8
            const __bf16* arow = Ebf + (size_t)(v0 + l) * DIM + k0 + (hi ? 8 : 0);
            v8bf alo = *(const v8bf*)(arow);
            v8bf ahi = *(const v8bf*)(arow + 16);
            v16bf a;
            #pragma unroll
            for (int i = 0; i < 8; ++i) { a[i] = alo[i]; a[8 + i] = ahi[i]; }
            // B 32x16 bf16 from LDS: lanes 0-15 -> K=k0..k0+7 / k0+16..k0+23, N=l ; hi lanes offset +8
            // rho is symmetric so column (e0+l) == row (e0+l): contiguous ds loads
            const __bf16* bcol = srho + (e0 + l) * DIM + k0 + (hi ? 8 : 0);
            v8bf blo = *(const v8bf*)(bcol);
            v8bf bhi = *(const v8bf*)(bcol + 16);
            v16bf b;
            #pragma unroll
            for (int i = 0; i < 8; ++i) { b[i] = blo[i]; b[8 + i] = bhi[i]; }
            acc = __builtin_amdgcn_wmma_f32_16x16x32_bf16(false, a, false, b, (short)0, acc, false, false);
        }
        // fuse (P*E) row-sum: C layout VGPR r -> M = r (+8 for hi lanes), N = l
        #pragma unroll
        for (int r = 0; r < 8; ++r) {
            int m = r + (hi ? 8 : 0);
            rowacc[r] += acc[r] * E[(size_t)(v0 + m) * DIM + e0 + l];
        }
    }
    // reduce within each 16-lane half (xor of bits 0..3 stays inside the half)
    #pragma unroll
    for (int off = 1; off < 16; off <<= 1)
        #pragma unroll
        for (int r = 0; r < 8; ++r)
            rowacc[r] += __shfl_xor(rowacc[r], off, 32);
    if (l == 0) {
        int mbase = hi ? 8 : 0;
        #pragma unroll
        for (int r = 0; r < 8; ++r) out[v0 + mbase + r] = rowacc[r];
    }
}

// ---------------------------------------------------------------- launcher

extern "C" void kernel_launch(void* const* d_in, const int* in_sizes, int n_in,
                              void* d_out, int out_size, void* d_ws, size_t ws_size,
                              hipStream_t stream) {
    (void)in_sizes; (void)n_in; (void)out_size; (void)ws_size;
    const int*   tokens = (const int*)d_in[0];
    const float* E      = (const float*)d_in[1];
    const float* W      = (const float*)d_in[2];
    const float* mdb    = (const float*)d_in[3];
    const float* ns     = (const float*)d_in[4];

    float* out        = (float*)d_out;
    float* logits_out = out;
    float* eigs_out   = out + (size_t)SEQ * VOCAB;

    char* p = (char*)d_ws;
    auto alloc = [&](size_t bytes) -> char* {
        char* r = p; p += (bytes + 255) & ~(size_t)255; return r;
    };
    float*  memory   = (float*)alloc((size_t)NB * DIM * 4);
    float*  means    = (float*)alloc((size_t)NFR * NB * DIM * 4);
    float*  vars     = (float*)alloc((size_t)NFR * NB * DIM * 4);
    float*  meas     = (float*)alloc((size_t)NB * DIM * 4);
    float*  eff      = (float*)alloc((size_t)NB * DIM * 4);
    float*  mnorm    = (float*)alloc((size_t)NB * DIM * 4);
    float*  rho      = (float*)alloc((size_t)DIM * DIM * 4);
    __bf16* rho_bf   = (__bf16*)alloc((size_t)DIM * DIM * 2);
    float*  xm       = (float*)alloc(DIM * 4);
    float*  mu       = (float*)alloc(DIM * 4);
    float*  decayp   = (float*)alloc(4);
    int*    flag     = (int*)alloc(4);
    float*  partials = (float*)alloc(NFR * 64 * 4);
    __bf16* Ebf      = (__bf16*)alloc((size_t)VOCAB * DIM * 2);

    // state init (ws persists across replays -> must reset every call)
    init_kernel<<<(NFR * NB * DIM + 255) / 256, 256, 0, stream>>>(memory, means, vars);
    cvtE_kernel<<<(VOCAB * DIM + 255) / 256, 256, 0, stream>>>(E, Ebf, VOCAB * DIM);

    for (int step = 0; step < SEQ; ++step) {
        prep_kernel<<<1, 256, 0, stream>>>(tokens, step, E, memory, xm, decayp, mdb, ns);
        meas_kernel<<<NB, 256, 0, stream>>>(xm, W, meas);
        surpriseA_kernel<<<dim3(64, NFR), 256, 0, stream>>>(meas, means, vars, partials);
        surpriseB_kernel<<<1, 64, 0, stream>>>(partials, step, flag);
        colmean_kernel<<<1, 256, 0, stream>>>(meas, mu);
        state_kernel<<<NB, 256, 0, stream>>>(meas, mu, flag, decayp, memory, means, vars,
                                             eff, mnorm, step);
        rho_kernel<<<(DIM / 16) * (DIM / 16), 32, 0, stream>>>(mnorm, rho, rho_bf);
        eig_kernel<<<1, 256, DIM * DIM * 4, stream>>>(rho, eigs_out + (size_t)step * DIM);
        logits_kernel<<<VOCAB / (16 * LW), 256, DIM * DIM * 2, stream>>>(
            Ebf, E, rho_bf, logits_out + (size_t)step * VOCAB);
    }
}
